// DeformableConvNorm_86466281603196
// MI455X (gfx1250) — matrix operand order repaired and checked by reading, not compiled, not run on hardware
//
#include <hip/hip_runtime.h>

// ---------------------------------------------------------------------------
// DCNv2 (modulated deformable conv 3x3, s=1, p=1) + GroupNorm(16), f32 I/O.
// GEMM per batch: out[256,4096] = W[256,2304] x sampled[2304,4096]
// K ordering k = kk*256 + c so each 32-wide K chunk has constant kk.
// Matrix path: v_wmma_f32_16x16x32_bf16, double-buffered LDS pipeline,
// 32-bit saddr-form gathers (uniform SGPR base + VGPR byte offset + imm).
// ---------------------------------------------------------------------------

typedef __attribute__((ext_vector_type(16))) __bf16        v16bf;
typedef __attribute__((ext_vector_type(8)))  float         v8f;
typedef __attribute__((ext_vector_type(4)))  unsigned int  u32x4;

union FragBF { v16bf v; u32x4 q[2]; };

__device__ __forceinline__ unsigned short f2bf(float f) {
  unsigned u = __float_as_uint(f);
  return (unsigned short)((u + 0x7FFFu + ((u >> 16) & 1u)) >> 16);
}
__device__ __forceinline__ unsigned packbf2(float a, float b) {
  return (unsigned)f2bf(a) | ((unsigned)f2bf(b) << 16);
}
__device__ __forceinline__ int iclamp(int v, int lo, int hi) {
  return v < lo ? lo : (v > hi ? hi : v);
}

#define WS_CONV   0u
#define WS_WBF    16777216u              // 4*256*4096 floats
#define WS_PART   (WS_WBF + 1179648u)    // 256*2304 bf16
#define WS_STATS  (WS_PART + 32768u)     // 4*64*32 floats of partials

// ---------------------------------------------------------------------------
// Kernel 0: weight f32 [O][C][KK] -> bf16 [m][kk*256 + c]
// ---------------------------------------------------------------------------
__global__ void dcn_prep(const float* __restrict__ wgt,
                         unsigned short* __restrict__ Wbf) {
  int t = blockIdx.x * 256 + threadIdx.x;
  if (t >= 256 * 2304) return;
  int m  = t / 2304;
  int q  = t - m * 2304;
  int kk = q >> 8;
  int c  = q & 255;
  Wbf[t] = f2bf(wgt[m * 2304 + c * 9 + kk]);
}

// ---------------------------------------------------------------------------
// Kernel 1: fused sampling + implicit GEMM via WMMA, software pipelined.
//   grid = (64 n-tiles, 4 batches), block = 256 threads (8 waves).
//   BM = 256 (all Cout), BN = 64, KC = 32, 72 K-chunks, ping-pong LDS.
// ---------------------------------------------------------------------------
__global__ __launch_bounds__(256)
void dcn_main(const float* __restrict__ xin,
              const float* __restrict__ off,
              const float* __restrict__ msk,
              const unsigned short* __restrict__ Wbf,
              float* __restrict__ convOut,
              float* __restrict__ partials) {
  __shared__ __attribute__((aligned(16))) unsigned short sA[2][256 * 40];
  __shared__ __attribute__((aligned(16))) unsigned short sB[2][64 * 40];
  __shared__ float sWred[64];

  const int t    = threadIdx.x;
  const int lane = t & 31;
  const int w    = t >> 5;
  const int b    = blockIdx.y;
  const int nTile0 = blockIdx.x * 64;
  const int m0   = (w >> 1) * 64;     // wave M offset (4 tiles of 16)
  const int n0   = (w & 1) * 32;      // wave N offset (2 tiles of 16)
  const int nl   = t & 63;            // this thread's spatial position
  const int cs   = t >> 6;            // this thread's 8-channel slice (0..3)
  const int n    = nTile0 + nl;
  const int ho   = n >> 6, wo = n & 63;

  // Uniform (SGPR) base for gathers; per-tap 32-bit byte offsets in VGPRs.
  const char* xbase = (const char*)(xin + ((size_t)b << 20));

  // per-kk gather state (registers, recomputed at kk boundaries)
  unsigned go0 = 0, go1 = 0, go2 = 0, go3 = 0;
  float    gw0 = 0, gw1 = 0, gw2 = 0, gw3 = 0;

  // staging registers for the next chunk (filled under current WMMAs)
  u32x4 aw0, aw1, aw2, aw3;
  float gv[8];

  auto calcParams = [&](int kk) {
    const float dy = off[((size_t)b * 18 + kk * 2)     * 4096 + n];
    const float dx = off[((size_t)b * 18 + kk * 2 + 1) * 4096 + n];
    const float mk = msk[((size_t)b * 9  + kk)         * 4096 + n];
    const float py = (float)(ho - 1 + kk / 3) + dy;
    const float px = (float)(wo - 1 + kk % 3) + dx;
    const float fy = floorf(py), fx = floorf(px);
    const int y0 = (int)fy, x0 = (int)fx;
    const int y1 = y0 + 1,  x1 = x0 + 1;
    const float ly = py - fy, lx = px - fx;
    const float hy = 1.f - ly, hx = 1.f - lx;
    const float vy0 = (y0 >= 0 && y0 < 64) ? 1.f : 0.f;
    const float vy1 = (y1 >= 0 && y1 < 64) ? 1.f : 0.f;
    const float vx0 = (x0 >= 0 && x0 < 64) ? 1.f : 0.f;
    const float vx1 = (x1 >= 0 && x1 < 64) ? 1.f : 0.f;
    const int cy0 = iclamp(y0, 0, 63), cy1 = iclamp(y1, 0, 63);
    const int cx0 = iclamp(x0, 0, 63), cx1 = iclamp(x1, 0, 63);
    const unsigned cb = (unsigned)cs * 131072u;  // cs*8*4096 floats, bytes
    go0 = cb + (unsigned)(cy0 * 64 + cx0) * 4u;
    go1 = cb + (unsigned)(cy0 * 64 + cx1) * 4u;
    go2 = cb + (unsigned)(cy1 * 64 + cx0) * 4u;
    go3 = cb + (unsigned)(cy1 * 64 + cx1) * 4u;
    gw0 = hy * hx * mk * vy0 * vx0;
    gw1 = hy * lx * mk * vy0 * vx1;
    gw2 = ly * hx * mk * vy1 * vx0;
    gw3 = ly * lx * mk * vy1 * vx1;
  };

  auto issueLoads = [&](int ch) {
    const int c0 = (ch & 7) * 32;
    if ((ch & 7) == 0) calcParams(ch >> 3);
    // A tile row for this thread: 32 bf16 of reordered weight (L2 resident)
    const char* wp = (const char*)Wbf + (unsigned)t * 4608u + (unsigned)ch * 64u;
    aw0 = *(const u32x4*)(wp);
    aw1 = *(const u32x4*)(wp + 16);
    aw2 = *(const u32x4*)(wp + 32);
    aw3 = *(const u32x4*)(wp + 48);
    // B tile: 8 channels x 1 position, 4 taps each; saddr + imm offset form
    const char* xc = xbase + (size_t)c0 * 16384;  // uniform
#pragma unroll
    for (int j = 0; j < 8; ++j) {
      const char* p = xc + (size_t)j * 16384;     // imm-foldable constant
      gv[j] = gw0 * *(const float*)(p + go0) +
              gw1 * *(const float*)(p + go1) +
              gw2 * *(const float*)(p + go2) +
              gw3 * *(const float*)(p + go3);
    }
  };

  auto storeTile = [&](int ch) {
    const int buf = ch & 1;
    unsigned short* a = &sA[buf][t * 40];
    *(u32x4*)(a)      = aw0;
    *(u32x4*)(a + 8)  = aw1;
    *(u32x4*)(a + 16) = aw2;
    *(u32x4*)(a + 24) = aw3;
    u32x4 pv = {packbf2(gv[0], gv[1]), packbf2(gv[2], gv[3]),
                packbf2(gv[4], gv[5]), packbf2(gv[6], gv[7])};
    *(u32x4*)(&sB[buf][nl * 40 + cs * 8]) = pv;
  };

  v8f acc[4][2] = {};

  // fragment LDS offsets (ushort units), fixed per lane
  const int kbB = (lane < 16) ? 0 : 8;    // B: K in {kbB..+7, kbB+16..+23}
  const int kbA = (lane < 16) ? 0 : 8;    // A: K in {kbA..+7, kbA+16..+23}
  const int bRow0 = (n0 + (lane & 15)) * 40 + kbB;
  const int bRow1 = (n0 + 16 + (lane & 15)) * 40 + kbB;
  const int aRowB = (m0 + (lane & 15)) * 40 + kbA;

  // prologue
  issueLoads(0);
  storeTile(0);
  __syncthreads();

  for (int ch = 0; ch < 72; ++ch) {
    if (ch < 71) issueLoads(ch + 1);  // gathers in flight under WMMAs

    const unsigned short* A = sA[ch & 1];
    const unsigned short* B = sB[ch & 1];
    FragBF fb0, fb1;
    fb0.q[0] = *(const u32x4*)(&B[bRow0]);
    fb0.q[1] = *(const u32x4*)(&B[bRow0 + 16]);
    fb1.q[0] = *(const u32x4*)(&B[bRow1]);
    fb1.q[1] = *(const u32x4*)(&B[bRow1 + 16]);
#pragma unroll
    for (int tm = 0; tm < 4; ++tm) {
      FragBF fa;
      const unsigned short* p = &A[aRowB + tm * 16 * 40];
      fa.q[0] = *(const u32x4*)(p);
      fa.q[1] = *(const u32x4*)(p + 16);
      acc[tm][0] = __builtin_amdgcn_wmma_f32_16x16x32_bf16(
          false, fa.v, false, fb0.v, (short)0, acc[tm][0], false, false);
      acc[tm][1] = __builtin_amdgcn_wmma_f32_16x16x32_bf16(
          false, fa.v, false, fb1.v, (short)0, acc[tm][1], false, false);
    }

    if (ch < 71) storeTile(ch + 1);   // write opposite buffer
    __syncthreads();
  }

  // --- epilogue: store conv output + deterministic group partial sums ------
  float* cOut = convOut + ((size_t)b << 20);
#pragma unroll
  for (int tm = 0; tm < 4; ++tm) {
    float s = 0.f, s2 = 0.f;
    const int mrow = m0 + tm * 16 + ((lane >> 4) << 3);  // + r below
#pragma unroll
    for (int tn = 0; tn < 2; ++tn) {
      const int ncol = nTile0 + n0 + tn * 16 + (lane & 15);
#pragma unroll
      for (int r = 0; r < 8; ++r) {
        const float v = acc[tm][tn][r];
        cOut[(size_t)(mrow + r) * 4096 + ncol] = v;
        s += v;
        s2 += v * v;
      }
    }
#pragma unroll
    for (int mshift = 16; mshift >= 1; mshift >>= 1) {
      s  += __shfl_xor(s,  mshift, 32);
      s2 += __shfl_xor(s2, mshift, 32);
    }
    if (lane == 0) {
      sWred[w * 8 + tm * 2]     = s;
      sWred[w * 8 + tm * 2 + 1] = s2;
    }
  }
  __syncthreads();
  if (t < 32) {
    const int g = t >> 1, j = t & 1;   // group = wave-pair wp*4 + tm
    const int wp = g >> 2, tm = g & 3;
    const float v = sWred[(wp * 2) * 8 + tm * 2 + j] +
                    sWred[(wp * 2 + 1) * 8 + tm * 2 + j];
    partials[((size_t)(b * 64 + blockIdx.x) * 32) + g * 2 + j] = v;
  }
}

// ---------------------------------------------------------------------------
// Kernel 2: reduce 64 n-tile partials per (b,g) -> mean, rsqrt(var+eps)
// ---------------------------------------------------------------------------
__global__ void dcn_stats(const float* __restrict__ partials,
                          float* __restrict__ statsF) {
  int i = threadIdx.x;
  if (i >= 64) return;
  const int b = i >> 4, g = i & 15;
  float s = 0.f, s2 = 0.f;
  for (int nt = 0; nt < 64; ++nt) {
    const float* p = partials + ((size_t)(b * 64 + nt) * 32) + g * 2;
    s  += p[0];
    s2 += p[1];
  }
  const float mean = s * (1.f / 65536.f);
  const float var  = s2 * (1.f / 65536.f) - mean * mean;
  statsF[2 * i]     = mean;
  statsF[2 * i + 1] = rsqrtf(var + 1e-5f);
}

// ---------------------------------------------------------------------------
// Kernel 3: apply GroupNorm, float4 vectorized.
// ---------------------------------------------------------------------------
__global__ __launch_bounds__(256)
void dcn_norm(const float* __restrict__ convOut,
              const float* __restrict__ statsF,
              const float* __restrict__ gamma,
              const float* __restrict__ beta,
              float* __restrict__ out) {
  const int t = blockIdx.x * 256 + threadIdx.x;  // 1,048,576 threads (x4 f32)
  const float4 v = ((const float4*)convOut)[t];
  const int m   = (t >> 10) & 255;
  const int b   = t >> 18;
  const int sid = b * 16 + (m >> 4);
  const float mean = statsF[2 * sid];
  const float rs   = statsF[2 * sid + 1];
  const float g    = gamma[m] * rs;
  const float be   = beta[m];
  float4 o;
  o.x = (v.x - mean) * g + be;
  o.y = (v.y - mean) * g + be;
  o.z = (v.z - mean) * g + be;
  o.w = (v.w - mean) * g + be;
  ((float4*)out)[t] = o;
}

// ---------------------------------------------------------------------------
extern "C" void kernel_launch(void* const* d_in, const int* in_sizes, int n_in,
                              void* d_out, int out_size, void* d_ws,
                              size_t ws_size, hipStream_t stream) {
  (void)in_sizes; (void)n_in; (void)out_size; (void)ws_size;
  const float* xin   = (const float*)d_in[0];
  const float* off   = (const float*)d_in[1];
  const float* msk   = (const float*)d_in[2];
  const float* wgt   = (const float*)d_in[3];
  const float* gamma = (const float*)d_in[4];
  const float* beta  = (const float*)d_in[5];
  float* out = (float*)d_out;

  char* ws = (char*)d_ws;
  float*          convOut  = (float*)(ws + WS_CONV);
  unsigned short* Wbf      = (unsigned short*)(ws + WS_WBF);
  float*          partials = (float*)(ws + WS_PART);
  float*          statsF   = (float*)(ws + WS_STATS);

  dcn_prep<<<2304, 256, 0, stream>>>(wgt, Wbf);
  dcn_main<<<dim3(64, 4), 256, 0, stream>>>(xin, off, msk, Wbf, convOut,
                                            partials);
  dcn_stats<<<1, 64, 0, stream>>>(partials, statsF);
  dcn_norm<<<4096, 256, 0, stream>>>(convOut, statsF, gamma, beta, out);
}
